// LocalSelfAttention_76742475645613
// MI455X (gfx1250) — matrix operand order repaired and checked by reading, not compiled, not run on hardware
//
#include <hip/hip_runtime.h>

#define B_DIM   4
#define S_DIM   8192
#define HID_DIM 1024
#define H_DIM   16
#define D_DIM   64
#define CHUNK   64
#define NCHUNK  (S_DIM / CHUNK)      /* 128 */
#define HEAD_ELEMS ((size_t)B_DIM * H_DIM * S_DIM * D_DIM) /* 33554432 */

typedef __attribute__((ext_vector_type(16))) __bf16 v16bf;
typedef __attribute__((ext_vector_type(2)))  __bf16 v2bf;
typedef __attribute__((ext_vector_type(2)))  float  v2f;
typedef __attribute__((ext_vector_type(8)))  float  v8f;

union Frag32B { v16bf v; uint4 q[2]; };

// Native f32 -> bf16 convert (hardware v_cvt path, RNE).
__device__ __forceinline__ unsigned short f2bf(float f) {
  __bf16 h = (__bf16)f;
  return __builtin_bit_cast(unsigned short, h);
}
// Packed pair convert: lowers to a single v_cvt_pk_bf16_f32, result already a dword.
__device__ __forceinline__ unsigned int f2bf2(float lo, float hi) {
  v2f  s = {lo, hi};
  v2bf p = __builtin_convertvector(s, v2bf);
  return __builtin_bit_cast(unsigned int, p);
}

// A fragment, bf16 16x32, from row-major [m][k] LDS tile with leading dim ld.
// ISA layout: lanes 0-15 hold row m=lane, K = {0..7, 16..23}; lanes 16-31 K = {8..15, 24..31}.
__device__ __forceinline__ v16bf load_frag_a(const unsigned short* base, int ld, int lane) {
  int m  = lane & 15;
  int kb = (lane >> 4) << 3;
  const unsigned short* p = base + m * ld + kb;
  Frag32B f;
  f.q[0] = *(const uint4*)(p);        // K = kb .. kb+7
  f.q[1] = *(const uint4*)(p + 16);   // K = kb+16 .. kb+23
  return f.v;
}

// B fragment, bf16 32x16, from [n][k] LDS tile (ld = K stride).
// ISA layout: lanes 0-15 hold col n=lane, K = {0..7, 16..23}; lanes 16-31 K = {8..15, 24..31}.
__device__ __forceinline__ v16bf load_frag_b(const unsigned short* base, int ld, int lane) {
  int n  = lane & 15;
  int ks = (lane >> 4) << 3;
  const unsigned short* p = base + n * ld + ks;
  Frag32B f;
  f.q[0] = *(const uint4*)(p);
  f.q[1] = *(const uint4*)(p + 16);
  return f.v;
}

__device__ __forceinline__ v8f wmma_bf16(v16bf a, v16bf b, v8f c) {
  return __builtin_amdgcn_wmma_f32_16x16x32_bf16(false, a, false, b, (short)0, c, false, false);
}

// ---------------------------------------------------------------------------
// Phase 1: fused Q/K/V projection. grid = (512, 16, 3), block = 256 (8 waves).
// Each block: 64(M) x 64(N) tile, K swept in 32-chunks through LDS with on-the-
// fly f32->bf16 conversion. Output written head-major [B,H,S,D] as bf16.
// ---------------------------------------------------------------------------
__global__ __launch_bounds__(256)
void qkv_gemm_kernel(const float* __restrict__ X,
                     const float* __restrict__ Wq,
                     const float* __restrict__ Wk,
                     const float* __restrict__ Wv,
                     unsigned short* __restrict__ qkv) {
  __shared__ unsigned short Xs[64 * 32];   // [m][k]
  __shared__ unsigned short Ws[64 * 32];   // [n][k] (transposed weight tile)

  const int   n0 = blockIdx.y * 64;
  const int   z  = blockIdx.z;
  const float* W = (z == 0) ? Wq : (z == 1 ? Wk : Wv);
  const float wscale = (z == 1) ? 0.125f : 1.0f;      // fold 1/sqrt(D) into Wk
  unsigned short* outb = qkv + (size_t)z * HEAD_ELEMS;

  const int tid  = threadIdx.x;
  const int lane = tid & 31;
  const int wave = tid >> 5;
  const int row0 = blockIdx.x * 64;

  const int mt  = wave & 3;          // 4 m-tiles of 16
  const int ntb = (wave >> 2) << 1;  // 2 n-tiles of 16 per wave

  v8f acc0 = {};
  v8f acc1 = {};

  const int xr = tid >> 2;           // 0..63
  const int xc = (tid & 3) << 3;     // 0,8,16,24

  for (int kk = 0; kk < HID_DIM; kk += 32) {
    __syncthreads();
    {
      const float* src = X + (size_t)(row0 + xr) * HID_DIM + kk + xc;
      float4 a0 = ((const float4*)src)[0];
      float4 a1 = ((const float4*)src)[1];
      unsigned int* dst = (unsigned int*)(Xs + xr * 32 + xc);
      dst[0] = f2bf2(a0.x, a0.y);
      dst[1] = f2bf2(a0.z, a0.w);
      dst[2] = f2bf2(a1.x, a1.y);
      dst[3] = f2bf2(a1.z, a1.w);
      if (kk + 32 < HID_DIM) __builtin_prefetch(src + 32, 0, 1);   // global_prefetch
    }
#pragma unroll
    for (int i = 0; i < 8; ++i) {
      int linear = tid + (i << 8);          // 0..2047
      int n = linear & 63;
      int k = linear >> 6;                  // 0..31
      Ws[n * 32 + k] = f2bf(W[(size_t)(kk + k) * HID_DIM + n0 + n] * wscale);
    }
    __syncthreads();

    v16bf a  = load_frag_a(Xs + mt * 16 * 32, 32, lane);
    v16bf b0 = load_frag_b(Ws + (ntb + 0) * 16 * 32, 32, lane);
    v16bf b1 = load_frag_b(Ws + (ntb + 1) * 16 * 32, 32, lane);
    acc0 = wmma_bf16(a, b0, acc0);
    acc1 = wmma_bf16(a, b1, acc1);
  }

  const int hi = lane >> 4;
  const int nl = lane & 15;
#pragma unroll
  for (int j = 0; j < 2; ++j) {
    v8f acc = j ? acc1 : acc0;
    int n_global = n0 + (ntb + j) * 16 + nl;
    int h = n_global >> 6;
    int d = n_global & 63;
#pragma unroll
    for (int r = 0; r < 8; ++r) {
      int m_global = row0 + mt * 16 + r + hi * 8;
      int bb = m_global >> 13;
      int ss = m_global & (S_DIM - 1);
      outb[(((size_t)bb * H_DIM + h) * S_DIM + ss) * D_DIM + d] = f2bf(acc[r]);
    }
  }
}

// ---------------------------------------------------------------------------
// Phase 2: local chunked attention. grid = B*H*nC = 8192 blocks, 256 threads.
// Key/value window = [prev chunk | cur chunk] (128 keys); causal mask handles
// the chunk-0 wraparound exactly like jnp.roll + mask in the reference.
// ---------------------------------------------------------------------------
#define VT_LD 136   /* padded V^T leading dim (bank-conflict break, keeps 16B align) */

__global__ __launch_bounds__(256)
void local_attn_kernel(const unsigned short* __restrict__ qb,
                       const unsigned short* __restrict__ kb,
                       const unsigned short* __restrict__ vb,
                       float* __restrict__ out) {
  extern __shared__ unsigned short smem[];
  float*          Sc = (float*)smem;               // 64*128 f32   (32 KB)
  unsigned short* Vt = smem + 16384;               // 64 x VT_LD   (17 KB)
  unsigned short* Qs = Vt + 64 * VT_LD;            // 64*64
  unsigned short* Ks = Qs + 64 * 64;               // 128*64
  unsigned short* Pr = Qs;                         // probs alias dead Q/K region

  const int blk = blockIdx.x;
  const int c = blk & (NCHUNK - 1);
  const int h = (blk >> 7) & (H_DIM - 1);
  const int b = blk >> 11;
  const int pc = (c + NCHUNK - 1) & (NCHUNK - 1);

  const size_t head_off = ((size_t)b * H_DIM + h) * S_DIM * D_DIM;
  const unsigned short* Qg = qb + head_off + (size_t)c  * CHUNK * D_DIM;
  const unsigned short* Kp = kb + head_off + (size_t)pc * CHUNK * D_DIM;
  const unsigned short* Kc = kb + head_off + (size_t)c  * CHUNK * D_DIM;
  const unsigned short* Vp = vb + head_off + (size_t)pc * CHUNK * D_DIM;
  const unsigned short* Vc = vb + head_off + (size_t)c  * CHUNK * D_DIM;

  const int tid  = threadIdx.x;
  const int lane = tid & 31;
  const int wave = tid >> 5;

  { // Q tile: 4096 bf16, 2x uint4 per thread
    int e = tid << 4;
    *(uint4*)(Qs + e)     = *(const uint4*)(Qg + e);
    *(uint4*)(Qs + e + 8) = *(const uint4*)(Qg + e + 8);
  }
#pragma unroll
  for (int i = 0; i < 4; ++i) {       // K window: 8192 bf16
    int e = (tid + (i << 8)) << 3;
    const unsigned short* src = (e < 4096) ? (Kp + e) : (Kc + (e - 4096));
    *(uint4*)(Ks + e) = *(const uint4*)src;
  }
#pragma unroll
  for (int i = 0; i < 32; ++i) {      // V^T: Vt[d][p] = V[p][d]
    int idx = tid + (i << 8);
    int p = idx >> 6;
    int d = idx & 63;
    unsigned short val = (p < 64) ? Vp[p * 64 + d] : Vc[(p - 64) * 64 + d];
    Vt[d * VT_LD + p] = val;
  }
  __syncthreads();

  const int qt  = wave >> 1;          // 4 query tiles of 16
  const int ktb = (wave & 1) << 2;    // 8 key tiles of 16, 4 per wave
  const int hi  = lane >> 4;
  const int nl  = lane & 15;

  // dots = Q @ K^T  (K-dim = D = 64 -> 2 WMMAs per 16x16 tile), mask, spill f32
#pragma unroll
  for (int kt = ktb; kt < ktb + 4; ++kt) {
    v8f acc = {};
#pragma unroll
    for (int dstep = 0; dstep < D_DIM; dstep += 32) {
      v16bf a  = load_frag_a(Qs + qt * 16 * 64 + dstep, 64, lane);
      v16bf bf = load_frag_b(Ks + kt * 16 * 64 + dstep, 64, lane);
      acc = wmma_bf16(a, bf, acc);
    }
    int ncol = kt * 16 + nl;
    int kg = (ncol < 64) ? (pc * CHUNK + ncol) : (c * CHUNK + ncol - 64);
#pragma unroll
    for (int r = 0; r < 8; ++r) {
      int m  = qt * 16 + r + hi * 8;
      int qg = c * CHUNK + m;
      float v = acc[r];
      if (qg < kg) v = -1.0e9f;       // causal mask (kills wrapped chunk-0 keys)
      Sc[m * 128 + ncol] = v;
    }
  }
  __syncthreads();

  if (tid < 64) {                     // f32 softmax, one row per thread
    float* row = Sc + tid * 128;
    float mx = -3.0e38f;
    for (int j = 0; j < 128; ++j) mx = fmaxf(mx, row[j]);
    float sum = 0.0f;
    for (int j = 0; j < 128; ++j) { float e = __expf(row[j] - mx); row[j] = e; sum += e; }
    float inv = 1.0f / sum;
    unsigned int* prow = (unsigned int*)(Pr + tid * 128);
    for (int j = 0; j < 64; ++j) prow[j] = f2bf2(row[2 * j] * inv, row[2 * j + 1] * inv);
  }
  __syncthreads();

  // out = P @ V  (K-dim = 128 -> 4 WMMAs per 16x16 tile)
  const int dtb = (wave & 1) << 1;    // 4 d-tiles of 16, 2 per wave
#pragma unroll
  for (int dt = dtb; dt < dtb + 2; ++dt) {
    v8f acc = {};
#pragma unroll
    for (int ks = 0; ks < 128; ks += 32) {
      v16bf a  = load_frag_a(Pr + qt * 16 * 128 + ks, 128, lane);
      v16bf bf = load_frag_b(Vt + dt * 16 * VT_LD + ks, VT_LD, lane);
      acc = wmma_bf16(a, bf, acc);
    }
#pragma unroll
    for (int r = 0; r < 8; ++r) {
      int m = qt * 16 + r + hi * 8;
      int d = dt * 16 + nl;
      out[((size_t)(b * S_DIM + c * CHUNK + m)) * HID_DIM + h * D_DIM + d] = acc[r];
    }
  }
}

// ---------------------------------------------------------------------------
extern "C" void kernel_launch(void* const* d_in, const int* in_sizes, int n_in,
                              void* d_out, int out_size, void* d_ws, size_t ws_size,
                              hipStream_t stream) {
  (void)in_sizes; (void)n_in; (void)out_size; (void)ws_size;
  const float* hs = (const float*)d_in[0];
  const float* Wq = (const float*)d_in[1];
  const float* Wk = (const float*)d_in[2];
  const float* Wv = (const float*)d_in[3];
  float* out = (float*)d_out;
  unsigned short* qkv = (unsigned short*)d_ws;   // q | k | v, bf16, [B,H,S,D] each

  dim3 g1(S_DIM * B_DIM / 64, HID_DIM / 64, 3);  // (512, 16, 3)
  qkv_gemm_kernel<<<g1, 256, 0, stream>>>(hs, Wq, Wk, Wv, qkv);

  size_t shmem = 64 * 128 * sizeof(float)
               + (64 * VT_LD + 64 * 64 + 128 * 64) * sizeof(unsigned short); // 74752 B
  local_attn_kernel<<<B_DIM * H_DIM * NCHUNK, 256, shmem, stream>>>(
      qkv, qkv + HEAD_ELEMS, qkv + 2 * HEAD_ELEMS, out);
}
